// MultiHeadAttention_47107201302971
// MI455X (gfx1250) — matrix operand order repaired and checked by reading, not compiled, not run on hardware
//
#include <hip/hip_runtime.h>
#include <hip/hip_bf16.h>

// ---------------------------------------------------------------------------
// MI455X (gfx1250) causal multi-head attention, bf16 WMMA + fp32 accumulate.
//   k1: QKV projection  (x fp32 -> Q/K/V bf16 workspace, [B*H, S, 64])
//   k2: flash attention (online softmax, causal)      -> attn bf16 [B,S,1024]
//       K tile staged via GLOBAL_LOAD_ASYNC_TO_LDS_B128 (ASYNCcnt path)
//   k3: output projection                              -> out fp32 [B,S,1024]
// All GEMMs use v_wmma_f32_16x16x32_bf16 (wave32).
// ---------------------------------------------------------------------------

typedef __attribute__((ext_vector_type(16))) __bf16 v16bf;
typedef __attribute__((ext_vector_type(8)))  float  v8f;

namespace {
constexpr int Bn = 4, Sn = 2048, DIN = 1024, Hn = 16, DKn = 64, DVHn = 64, DMODELn = 1024;
constexpr float SCALE  = 0.125f;     // 1/sqrt(64)
constexpr float NEGBIG = -1.0e30f;
}

union AFrag { v16bf v; unsigned int u[8]; unsigned short h[16]; };

__device__ inline unsigned short f2bf(float f) {
  unsigned int u = __float_as_uint(f);
  u += 0x7FFFu + ((u >> 16) & 1u);          // round-to-nearest-even
  return (unsigned short)(u >> 16);
}

// Generic LDS pointer = {aperture_hi32, lds_offset_lo32} -> truncate.
__device__ inline unsigned lds_off(const void* p) {
  return (unsigned)(unsigned long long)p;
}

__device__ inline v8f wmma_bf16(const AFrag& a, const AFrag& b, v8f c) {
  // (neg_a, A, neg_b, B, c_mod, C, reuse_a, reuse_b)
  return __builtin_amdgcn_wmma_f32_16x16x32_bf16(false, a.v, false, b.v,
                                                 (short)0, c, false, false);
}

// ---------------------------------------------------------------------------
// Kernel 1: per-head QKV projection. grid = (S/64, B*H, 3), block = 128.
// ---------------------------------------------------------------------------
__global__ void __launch_bounds__(128)
qkv_proj_kernel(const float* __restrict__ x,
                const float* __restrict__ Wq,
                const float* __restrict__ Wk,
                const float* __restrict__ Wv,
                unsigned short* __restrict__ Qb,
                unsigned short* __restrict__ Kb,
                unsigned short* __restrict__ Vb) {
  __shared__ unsigned short xs[64][40];    // 64 rows x 32 K  (bf16, padded)
  __shared__ unsigned short wsT[64][40];   // 64 N   x 32 K  (transposed weight tile)

  const int st    = blockIdx.x;
  const int bh    = blockIdx.y;
  const int which = blockIdx.z;
  const int tid   = threadIdx.x;
  const int wave  = tid >> 5, lane = tid & 31;
  const int hf    = (lane >> 4) & 1, l16 = lane & 15;
  const int h     = bh % Hn;

  const float* W = (which == 0) ? Wq : (which == 1) ? Wk : Wv;
  W += (size_t)h * DIN * DKn;
  unsigned short* Out = (which == 0) ? Qb : (which == 1) ? Kb : Vb;

  const float* xb = x + ((size_t)(bh / Hn) * Sn + (size_t)st * 64) * DIN;

  v8f acc[4];
  #pragma unroll
  for (int nt = 0; nt < 4; ++nt)
    #pragma unroll
    for (int v = 0; v < 8; ++v) acc[nt][v] = 0.0f;

  for (int kk = 0; kk < DIN; kk += 32) {
    #pragma unroll
    for (int i = 0; i < 16; ++i) {               // stage x tile (coalesced)
      int e = tid + i * 128;
      int r = e >> 5, c = e & 31;
      xs[r][c] = f2bf(xb[(size_t)r * DIN + kk + c]);
    }
    #pragma unroll
    for (int i = 0; i < 16; ++i) {               // stage W tile, transposed
      int e = tid + i * 128;
      int k = e >> 6, n = e & 63;
      wsT[n][k] = f2bf(W[(size_t)(kk + k) * DKn + n]);
    }
    __syncthreads();

    AFrag a;                                     // A: 16x32, rows = wave slice
    const int r = wave * 16 + l16;
    #pragma unroll
    for (int v = 0; v < 8; ++v) {
      int k = (v >> 2) * 16 + hf * 8 + (v & 3) * 2;
      a.u[v] = *(const unsigned int*)&xs[r][k];
    }
    #pragma unroll
    for (int nt = 0; nt < 4; ++nt) {
      AFrag bfr;                                 // B: 32x16, col = lane
      const int n = nt * 16 + l16;
      #pragma unroll
      for (int v = 0; v < 8; ++v) {
        int k = (v >> 2) * 16 + hf * 8 + (v & 3) * 2;
        bfr.u[v] = *(const unsigned int*)&wsT[n][k];
      }
      acc[nt] = wmma_bf16(a, bfr, acc[nt]);
    }
    __syncthreads();
  }

  #pragma unroll
  for (int nt = 0; nt < 4; ++nt)
    #pragma unroll
    for (int v = 0; v < 8; ++v) {
      int srow = st * 64 + wave * 16 + v + hf * 8;
      int n    = nt * 16 + l16;
      Out[((size_t)bh * Sn + srow) * DKn + n] = f2bf(acc[nt][v]);
    }
}

// ---------------------------------------------------------------------------
// Kernel 2: causal flash attention. grid = (S/64, B*H), block = 128.
// K tile: async-copied global->LDS (row-padded) via the ASYNCcnt engine.
// V tile: transposed-staged through VGPRs. P relayout via per-wave LDS slab.
// ---------------------------------------------------------------------------
__global__ void __launch_bounds__(128)
flash_attn_kernel(const unsigned short* __restrict__ Qb,
                  const unsigned short* __restrict__ Kb,
                  const unsigned short* __restrict__ Vb,
                  unsigned short* __restrict__ Ab) {
  __shared__ __align__(16) unsigned short ksh[64][80]; // K tile, 160B rows (16B aligned)
  __shared__ unsigned short vT[64][72];                // V tile transposed: [n][kv]
  __shared__ unsigned short ps[4][16][72];             // per-wave P slab (C->A relayout)

  const int it   = blockIdx.x;
  const int bh   = blockIdx.y;
  const int b    = bh / Hn, h = bh % Hn;
  const int tid  = threadIdx.x;
  const int wave = tid >> 5, lane = tid & 31;
  const int hf   = (lane >> 4) & 1, l16 = lane & 15;
  const int s0   = it * 64;

  // Preload this wave's Q fragments (16 rows x 64 K  -> two 16x32 A-frags).
  AFrag qf[2];
  {
    const int row = s0 + wave * 16 + l16;
    const unsigned int* qp =
        (const unsigned int*)(Qb + ((size_t)bh * Sn + row) * DKn);
    #pragma unroll
    for (int kb = 0; kb < 2; ++kb)
      #pragma unroll
      for (int v = 0; v < 8; ++v) {
        int k = kb * 32 + (v >> 2) * 16 + hf * 8 + (v & 3) * 2;
        qf[kb].u[v] = qp[k >> 1];
      }
  }

  v8f o[4];
  float m_i[8], l_i[8];
  #pragma unroll
  for (int nt = 0; nt < 4; ++nt)
    #pragma unroll
    for (int v = 0; v < 8; ++v) o[nt][v] = 0.0f;
  #pragma unroll
  for (int v = 0; v < 8; ++v) { m_i[v] = -3.0e38f; l_i[v] = 0.0f; }

  const unsigned kshBase = lds_off(&ksh[0][0]);

  for (int jt = 0; jt <= it; ++jt) {
    const int j0 = jt * 64;
    __syncthreads();                             // prior-iter LDS reads done

    // --- async-copy K tile (64 rows x 128B) global -> LDS, 16B/lane chunks,
    //     row-padded to 160B during the copy (per-lane LDS dest address).
    {
      const unsigned long long gBase =
          (unsigned long long)(const void*)(Kb + ((size_t)bh * Sn + j0) * DKn);
      #pragma unroll
      for (int i = 0; i < 4; ++i) {
        int chunk = tid + i * 128;               // 512 chunks of 16B
        int row = chunk >> 3, seg = chunk & 7;
        unsigned gOff = (unsigned)(row * 128 + seg * 16);
        unsigned lOff = kshBase + (unsigned)(row * 160 + seg * 16);
        asm volatile("global_load_async_to_lds_b128 %0, %1, %2"
                     :: "v"(lOff), "v"(gOff), "s"(gBase) : "memory");
      }
    }

    #pragma unroll
    for (int i = 0; i < 32; ++i) {               // stage V tile transposed
      int e = tid + i * 128;
      int k = e >> 6, n = e & 63;
      vT[n][k] = Vb[((size_t)bh * Sn + j0 + k) * DVHn + n];
    }
    if (jt + 1 <= it)                            // prefetch next K tile into GL2
      __builtin_prefetch(Kb + ((size_t)bh * Sn + j0 + 64) * DKn + lane * 4, 0, 1);

    asm volatile("s_wait_asynccnt 0" ::: "memory");
    __syncthreads();

    // --- scores: S = Q * K^T  (K B-frags from LDS: dword pair reads)
    v8f sc[4];
    #pragma unroll
    for (int nt = 0; nt < 4; ++nt)
      #pragma unroll
      for (int v = 0; v < 8; ++v) sc[nt][v] = 0.0f;

    #pragma unroll
    for (int kb = 0; kb < 2; ++kb)
      #pragma unroll
      for (int nt = 0; nt < 4; ++nt) {
        AFrag bk;
        const int nrow = nt * 16 + l16;
        #pragma unroll
        for (int v = 0; v < 8; ++v) {
          int k = kb * 32 + (v >> 2) * 16 + hf * 8 + (v & 3) * 2;
          bk.u[v] = *(const unsigned int*)&ksh[nrow][k];
        }
        sc[nt] = wmma_bf16(qf[kb], bk, sc[nt]);
      }

    // --- scale + causal mask (diagonal tile only)
    #pragma unroll
    for (int nt = 0; nt < 4; ++nt)
      #pragma unroll
      for (int v = 0; v < 8; ++v) {
        float val = sc[nt][v] * SCALE;
        if (jt == it) {
          int col = nt * 16 + l16;
          int row = wave * 16 + v + hf * 8;
          if (col > row) val = NEGBIG;
        }
        sc[nt][v] = val;
      }

    // --- online softmax: row stats over the 16-lane half-group
    float alpha[8], rs[8];
    #pragma unroll
    for (int v = 0; v < 8; ++v) {
      float t = fmaxf(fmaxf(sc[0][v], sc[1][v]), fmaxf(sc[2][v], sc[3][v]));
      t = fmaxf(t, __shfl_xor(t, 1));
      t = fmaxf(t, __shfl_xor(t, 2));
      t = fmaxf(t, __shfl_xor(t, 4));
      t = fmaxf(t, __shfl_xor(t, 8));
      float mn = fmaxf(m_i[v], t);
      alpha[v] = __expf(m_i[v] - mn);
      m_i[v]   = mn;
      rs[v]    = 0.0f;
    }

    #pragma unroll
    for (int nt = 0; nt < 4; ++nt)
      #pragma unroll
      for (int v = 0; v < 8; ++v) {
        float e = __expf(sc[nt][v] - m_i[v]);
        rs[v] += e;
        ps[wave][v + hf * 8][nt * 16 + l16] = f2bf(e);   // C-layout -> LDS
      }

    #pragma unroll
    for (int v = 0; v < 8; ++v) {
      float r = rs[v];
      r += __shfl_xor(r, 1);
      r += __shfl_xor(r, 2);
      r += __shfl_xor(r, 4);
      r += __shfl_xor(r, 8);
      l_i[v] = l_i[v] * alpha[v] + r;
    }
    #pragma unroll
    for (int nt = 0; nt < 4; ++nt)
      #pragma unroll
      for (int v = 0; v < 8; ++v) o[nt][v] *= alpha[v];

    __syncthreads();                             // ps visible for A-frag reads

    // --- O += P * V
    #pragma unroll
    for (int kb = 0; kb < 2; ++kb) {
      AFrag ap;                                  // reload P in A-layout
      #pragma unroll
      for (int v = 0; v < 8; ++v) {
        int k = kb * 32 + (v >> 2) * 16 + hf * 8 + (v & 3) * 2;
        ap.u[v] = *(const unsigned int*)&ps[wave][l16][k];
      }
      #pragma unroll
      for (int nt = 0; nt < 4; ++nt) {
        AFrag bv;
        const int n = nt * 16 + l16;
        #pragma unroll
        for (int v = 0; v < 8; ++v) {
          int k = kb * 32 + (v >> 2) * 16 + hf * 8 + (v & 3) * 2;
          bv.u[v] = *(const unsigned int*)&vT[n][k];
        }
        o[nt] = wmma_bf16(ap, bv, o[nt]);
      }
    }
  }

  // normalize and store concat-head layout [B, S, H*DVH] (bf16)
  #pragma unroll
  for (int v = 0; v < 8; ++v) {
    float inv = 1.0f / l_i[v];
    int srow  = s0 + wave * 16 + v + hf * 8;
    #pragma unroll
    for (int nt = 0; nt < 4; ++nt) {
      int col = nt * 16 + l16;
      Ab[(((size_t)b * Sn + srow) * Hn + h) * DVHn + col] = f2bf(o[nt][v] * inv);
    }
  }
}

// ---------------------------------------------------------------------------
// Kernel 3: output projection [B*S,1024] x [1024,1024]. grid=(128,16), 128thr.
// ---------------------------------------------------------------------------
__global__ void __launch_bounds__(128)
out_proj_kernel(const unsigned short* __restrict__ Ain,
                const float* __restrict__ Wo,
                float* __restrict__ out) {
  __shared__ unsigned short woT[64][40];         // [n][k] transposed Wo tile

  const int r0   = blockIdx.x * 64;
  const int n0   = blockIdx.y * 64;
  const int tid  = threadIdx.x;
  const int wave = tid >> 5, lane = tid & 31;
  const int hf   = (lane >> 4) & 1, l16 = lane & 15;

  v8f acc[4];
  #pragma unroll
  for (int nt = 0; nt < 4; ++nt)
    #pragma unroll
    for (int v = 0; v < 8; ++v) acc[nt][v] = 0.0f;

  const int row = r0 + wave * 16 + l16;
  const unsigned int* arow = (const unsigned int*)(Ain + (size_t)row * DMODELn);

  for (int kk = 0; kk < DMODELn; kk += 32) {
    #pragma unroll
    for (int i = 0; i < 16; ++i) {
      int e = tid + i * 128;
      int k = e >> 6, n = e & 63;
      woT[n][k] = f2bf(Wo[(size_t)(kk + k) * DMODELn + n0 + n]);
    }
    __syncthreads();

    AFrag a;                                     // A straight from bf16 ws
    #pragma unroll
    for (int v = 0; v < 8; ++v) {
      int k = kk + (v >> 2) * 16 + hf * 8 + (v & 3) * 2;
      a.u[v] = arow[k >> 1];
    }
    #pragma unroll
    for (int nt = 0; nt < 4; ++nt) {
      AFrag bw;
      const int n = nt * 16 + l16;
      #pragma unroll
      for (int v = 0; v < 8; ++v) {
        int k = (v >> 2) * 16 + hf * 8 + (v & 3) * 2;
        bw.u[v] = *(const unsigned int*)&woT[n][k];
      }
      acc[nt] = wmma_bf16(a, bw, acc[nt]);
    }
    __syncthreads();
  }

  #pragma unroll
  for (int nt = 0; nt < 4; ++nt)
    #pragma unroll
    for (int v = 0; v < 8; ++v) {
      int srow = r0 + wave * 16 + v + hf * 8;
      out[(size_t)srow * DMODELn + n0 + nt * 16 + l16] = acc[nt][v];
    }
}

// ---------------------------------------------------------------------------
extern "C" void kernel_launch(void* const* d_in, const int* in_sizes, int n_in,
                              void* d_out, int out_size, void* d_ws, size_t ws_size,
                              hipStream_t stream) {
  (void)in_sizes; (void)n_in; (void)out_size; (void)ws_size;

  const float* x  = (const float*)d_in[0];
  const float* Wq = (const float*)d_in[1];
  const float* Wk = (const float*)d_in[2];
  const float* Wv = (const float*)d_in[3];
  const float* Wo = (const float*)d_in[4];
  float* out = (float*)d_out;

  // workspace: Q | K | V | attn-out, all bf16 (4 x 16 MB = 64 MB)
  const size_t qkvElems = (size_t)Bn * Hn * Sn * DKn;
  unsigned short* Qb = (unsigned short*)d_ws;
  unsigned short* Kb = Qb + qkvElems;
  unsigned short* Vb = Kb + qkvElems;
  unsigned short* Ab = Vb + qkvElems;

  qkv_proj_kernel<<<dim3(Sn / 64, Bn * Hn, 3), 128, 0, stream>>>(
      x, Wq, Wk, Wv, Qb, Kb, Vb);
  flash_attn_kernel<<<dim3(Sn / 64, Bn * Hn), 128, 0, stream>>>(
      Qb, Kb, Vb, Ab);
  out_proj_kernel<<<dim3((Bn * Sn) / 64, DMODELn / 64), 128, 0, stream>>>(
      Ab, Wo, out);
}
